// ARC_ActionDecoder_17343077941664
// MI455X (gfx1250) — matrix operand order repaired and checked by reading, not compiled, not run on hardware
//
#include <hip/hip_runtime.h>

// ---------------------------------------------------------------------------
// VQ nearest-neighbor: argmin_j (||c_j||^2 - 2 e.c_j) over 1024 codebook rows
// for 65536 query vectors of dim 64.  Compute-bound GEMM -> bf16 WMMA with
// hi/lo split (bf16x3) for ~f32 accuracy; codebook double-buffered in LDS via
// async global->LDS copies (ASYNCcnt) so staging overlaps WMMA compute.
// ---------------------------------------------------------------------------

typedef __attribute__((ext_vector_type(16))) __bf16 v16bf;
typedef __attribute__((ext_vector_type(8)))  __bf16 v8bf;
typedef __attribute__((ext_vector_type(8)))  float  v8f;
typedef __attribute__((ext_vector_type(4)))  float  v4f;
typedef __attribute__((ext_vector_type(4)))  int    v4i;
typedef __attribute__((address_space(1)))    v4i*   as1_v4i;  // global
typedef __attribute__((address_space(3)))    v4i*   as3_v4i;  // LDS

#define DIMK   64      // embedding / codebook dim
#define NCODE  1024    // codebook entries
#define NQ     65536   // B*S query vectors
#define CHUNK  128     // codebook rows staged in LDS per pass
#define NCHUNK (NCODE / CHUNK)
#define LSTR   72      // LDS row stride in bf16 elems (64 + 8 pad -> 144B,
                       // makes the 16B fragment reads bank-conflict free)

#if __has_builtin(__builtin_amdgcn_global_load_async_to_lds_b128)
#define HAVE_ASYNC 1
#else
#define HAVE_ASYNC 0
#endif

// --- prep: split codebook f32 -> bf16 hi/lo, compute ||c||^2 ----------------
__global__ void vq_prep(const float* __restrict__ cb,
                        unsigned short* __restrict__ hi,
                        unsigned short* __restrict__ lo,
                        float* __restrict__ sq) {
    const int row = blockIdx.x;       // one row per block, 64 threads
    const int t   = threadIdx.x;
    float c  = cb[row * DIMK + t];
    __bf16 h = (__bf16)c;                      // RNE
    __bf16 l = (__bf16)(c - (float)h);         // residual
    hi[row * DIMK + t] = __builtin_bit_cast(unsigned short, h);
    lo[row * DIMK + t] = __builtin_bit_cast(unsigned short, l);
    float p = c * c;
    #pragma unroll
    for (int o = 16; o > 0; o >>= 1) p += __shfl_xor(p, o, 32);
    __shared__ float s2[2];
    if ((t & 31) == 0) s2[t >> 5] = p;
    __syncthreads();
    if (t == 0) sq[row] = s2[0] + s2[1];
}

// --- chunk staging: 128 rows x 64 bf16 (hi+lo) into padded LDS tile ---------
// 1024 16B groups per array; 256 threads x 4 groups each.
__device__ __forceinline__ void stage_chunk(const unsigned short* __restrict__ gH,
                                            const unsigned short* __restrict__ gL,
                                            __bf16* dH, __bf16* dL, int t) {
    unsigned short* gh = const_cast<unsigned short*>(gH);
    unsigned short* gl = const_cast<unsigned short*>(gL);
    #pragma unroll
    for (int i = 0; i < 4; ++i) {
        const int g = t + i * 256;
        const int r = g >> 3;            // row 0..127
        const int s = (g & 7) * 8;       // 8-elem (16B) group within row
#if HAVE_ASYNC
        __builtin_amdgcn_global_load_async_to_lds_b128(
            (as1_v4i)(gh + r * DIMK + s),
            (as3_v4i)(dH + r * LSTR + s), 0, 0);
        __builtin_amdgcn_global_load_async_to_lds_b128(
            (as1_v4i)(gl + r * DIMK + s),
            (as3_v4i)(dL + r * LSTR + s), 0, 0);
#else
        *(uint4*)(dH + r * LSTR + s) = *(const uint4*)(gh + r * DIMK + s);
        *(uint4*)(dL + r * LSTR + s) = *(const uint4*)(gl + r * DIMK + s);
#endif
    }
}

__device__ __forceinline__ void wait_async() {
#if HAVE_ASYNC
#if __has_builtin(__builtin_amdgcn_s_wait_asynccnt)
    __builtin_amdgcn_s_wait_asynccnt(0);
#else
    asm volatile("s_wait_asynccnt 0x0" ::: "memory");
#endif
#endif
}

// --- main: per wave a 16-query stripe, loop all codebook tiles --------------
__global__ __launch_bounds__(256) void vq_argmin(
        const float* __restrict__ emb,
        const unsigned short* __restrict__ cbh_u,
        const unsigned short* __restrict__ cbl_u,
        const float* __restrict__ csq,
        float* __restrict__ out) {
    __shared__ __attribute__((aligned(16))) __bf16 sH[2][CHUNK * LSTR];
    __shared__ __attribute__((aligned(16))) __bf16 sL[2][CHUNK * LSTR];
    __shared__ __attribute__((aligned(16))) float  sQ[NCODE];

    const int t    = threadIdx.x;
    const int lane = t & 31;
    const int half = lane >> 4;   // half-wave id (ISA fragment layouts)
    const int l16  = lane & 15;
    const int wave = t >> 5;
    const int q    = (blockIdx.x * 8 + wave) * 16 + l16;  // this lane's query

    // ---- stage all ||c||^2 (4KB) once; kick off async stage of chunk 0
    #pragma unroll
    for (int i = 0; i < NCODE / 256; ++i) sQ[t + i * 256] = csq[t + i * 256];
    stage_chunk(cbh_u, cbl_u, sH[0], sL[0], t);

    // ---- B fragments (K x 16 queries), kept resident for whole kernel.
    // B 32x16 bf16 layout: lane holds column N=l16, two 8-elem K runs at
    // kstep*32 + half*8 and +16.  Split f32 -> hi/lo bf16 in registers.
    v16bf bh[2], bl[2];
    {
        const float* erow = emb + (size_t)q * DIMK;
        #pragma unroll
        for (int ks = 0; ks < 2; ++ks) {
            const v4f* e4 = (const v4f*)(erow + ks * 32 + half * 8);
            #pragma unroll
            for (int v = 0; v < 4; ++v) {
                v4f x = e4[(v & 1) + (v >> 1) * 4];
                #pragma unroll
                for (int c = 0; c < 4; ++c) {
                    float  f = x[c];
                    __bf16 h = (__bf16)f;
                    bh[ks][v * 4 + c] = h;
                    bl[ks][v * 4 + c] = (__bf16)(f - (float)h);
                }
            }
        }
    }

    float bestv = 3.402823466e38f;
    int   besti = 0;

    for (int ch = 0; ch < NCHUNK; ++ch) {
        const int cur = ch & 1;
        wait_async();        // my async stage of chunk `ch` landed in LDS
        __syncthreads();     // ... for every wave; prev compute on this buf done
        if (ch + 1 < NCHUNK)  // background-stage next chunk into other buffer
            stage_chunk(cbh_u + (size_t)(ch + 1) * CHUNK * DIMK,
                        cbl_u + (size_t)(ch + 1) * CHUNK * DIMK,
                        sH[cur ^ 1], sL[cur ^ 1], t);

        const int cbase = ch * CHUNK;
        // ---- 8 codebook tiles of 16 rows each
        #pragma unroll
        for (int mt = 0; mt < CHUNK / 16; ++mt) {
            const __bf16* rH = sH[cur] + (mt * 16 + l16) * LSTR;
            const __bf16* rL = sL[cur] + (mt * 16 + l16) * LSTR;
            v8f acc = {};
            #pragma unroll
            for (int ks = 0; ks < 2; ++ks) {
                // A 16x32 bf16 layout: lane row M=l16, two 8-elem K runs at
                // e0 = ks*32 + half*8 and e0+16.
                const int e0 = ks * 32 + half * 8;
                v8bf ah0 = *(const v8bf*)(rH + e0);
                v8bf ah1 = *(const v8bf*)(rH + e0 + 16);
                v8bf al0 = *(const v8bf*)(rL + e0);
                v8bf al1 = *(const v8bf*)(rL + e0 + 16);
                v16bf ah = __builtin_shufflevector(ah0, ah1,
                    0,1,2,3,4,5,6,7,8,9,10,11,12,13,14,15);
                v16bf al = __builtin_shufflevector(al0, al1,
                    0,1,2,3,4,5,6,7,8,9,10,11,12,13,14,15);
                // bf16x3: hi*hi + hi*lo + lo*hi, single f32 accumulator
                acc = __builtin_amdgcn_wmma_f32_16x16x32_bf16(
                        false, ah, false, bh[ks], (short)0, acc, false, false);
                acc = __builtin_amdgcn_wmma_f32_16x16x32_bf16(
                        false, ah, false, bl[ks], (short)0, acc, false, false);
                acc = __builtin_amdgcn_wmma_f32_16x16x32_bf16(
                        false, al, false, bh[ks], (short)0, acc, false, false);
            }
            // C layout: lane holds query N=l16, rows M = half*8 + r (r=0..7)
            const int jbase = cbase + mt * 16 + half * 8;
            v4f q0 = *(const v4f*)(sQ + jbase);
            v4f q1 = *(const v4f*)(sQ + jbase + 4);
            #pragma unroll
            for (int r = 0; r < 8; ++r) {
                const float cq = (r < 4) ? q0[r] : q1[r - 4];
                const float s  = cq - 2.0f * acc[r];
                if (s < bestv) { bestv = s; besti = jbase + r; }  // strict < => lowest j on ties
            }
        }
        __syncthreads();   // compute on buf `cur` done before it is re-staged
    }

    // combine the two half-wave row ranges for the same query
    const float ov = __shfl_xor(bestv, 16, 32);
    const int   oi = __shfl_xor(besti, 16, 32);
    if (ov < bestv || (ov == bestv && oi < besti)) besti = oi;
    if (half == 0) out[q] = (float)besti;   // idx < 1024 -> exact in f32
}

// ---------------------------------------------------------------------------
extern "C" void kernel_launch(void* const* d_in, const int* in_sizes, int n_in,
                              void* d_out, int out_size, void* d_ws, size_t ws_size,
                              hipStream_t stream) {
    const float* emb = (const float*)d_in[0];   // (16,4096,64) f32
    const float* cb  = (const float*)d_in[1];   // (1024,64)    f32
    float* out = (float*)d_out;                 // 65536 indices as f32

    char* ws = (char*)d_ws;                     // 260KB scratch layout:
    unsigned short* cbh = (unsigned short*)(ws);                              // 128KB hi
    unsigned short* cbl = (unsigned short*)(ws + (size_t)NCODE * DIMK * 2);   // 128KB lo
    float*          csq = (float*)(ws + (size_t)NCODE * DIMK * 4);            // 4KB ||c||^2

    vq_prep<<<NCODE, DIMK, 0, stream>>>(cb, cbh, cbl, csq);
    vq_argmin<<<NQ / 128, 256, 0, stream>>>(emb, cbh, cbl, csq, out);
}
